// CausalSelfAttention_36180804502127
// MI455X (gfx1250) — matrix operand order repaired and checked by reading, not compiled, not run on hardware
//
#include <hip/hip_runtime.h>

// ---------------- Problem constants (from reference) ----------------
#define B_  2
#define T_  2048
#define C_  2048
#define H_  16
#define HD_ 128
#define C3_ 6144
#define BT_ (B_ * T_)

typedef __attribute__((ext_vector_type(16))) __bf16 v16bf;
typedef __attribute__((ext_vector_type(8)))  float  v8f;
typedef unsigned int v4u __attribute__((ext_vector_type(4)));
typedef int          v8i __attribute__((ext_vector_type(8)));
typedef int          v4i __attribute__((ext_vector_type(4)));

union FragBF {
    v16bf bf;
    uint4 q[2];
    unsigned short u[16];
};

__device__ __forceinline__ unsigned short f2bf(float f) {
    unsigned int u = __float_as_uint(f);
    unsigned int r = u + 0x7FFFu + ((u >> 16) & 1u);   // round-to-nearest-even
    return (unsigned short)(r >> 16);
}

// ---------------- Tensor Data Mover: 2D bf16 tile -> LDS ----------------
// Loads a [tile_rows x 32] bf16 tile (rows contiguous along K) into LDS at
// lds_off, row-major packed [tile_rows][32]. Descriptor per ISA 8.3/8.4.
__device__ __forceinline__ void tdm_load_tile(unsigned lds_off,
                                              const unsigned short* gptr,
                                              unsigned tile_rows,
                                              unsigned tensor_rows,
                                              unsigned rowlen /*K elements*/) {
    unsigned long long ga = (unsigned long long)(const void*)gptr;
    v4u g0;
    g0[0] = 1u;                                    // count=1, user-mode, no gather
    g0[1] = lds_off;                               // lds_addr (bytes)
    g0[2] = (unsigned)(ga & 0xFFFFFFFFu);          // global_addr[31:0]
    g0[3] = (unsigned)((ga >> 32) & 0x01FFFFFFu)   // global_addr[56:32]
          | (2u << 30);                            // type = 2 ("image")
    v8i g1;
    g1[0] = (int)(1u << 16);                       // wg_mask=0, data_size=1 (2 B)
    g1[1] = (int)((rowlen & 0xFFFFu) << 16);       // tensor_dim0[15:0] @ bits 63:48
    g1[2] = (int)(((rowlen >> 16) & 0xFFFFu)       // tensor_dim0[31:16]
          | ((tensor_rows & 0xFFFFu) << 16));      // tensor_dim1[15:0]
    g1[3] = (int)(((tensor_rows >> 16) & 0xFFFFu)  // tensor_dim1[31:16]
          | (32u << 16));                          // tile_dim0 = 32
    g1[4] = (int)(tile_rows & 0xFFFFu);            // tile_dim1, tile_dim2=0
    g1[5] = (int)rowlen;                           // tensor_dim0_stride[31:0]
    g1[6] = 0;                                     // stride hi / dim1_stride lo
    g1[7] = 0;
    v4i z4 = {0, 0, 0, 0};
    v8i z8 = {0, 0, 0, 0, 0, 0, 0, 0};
    // clang-23 / therock-10.0 form: (g0, g1, g2, g3, extra, cpol)
    __builtin_amdgcn_tensor_load_to_lds(g0, g1, z4, z4, z8, 0);
}

// ---------------- elementwise casts ----------------
__global__ __launch_bounds__(256)
void cast_bf16_kernel(const float* __restrict__ in, unsigned short* __restrict__ out, int n) {
    int i = blockIdx.x * 256 + threadIdx.x;
    if (i < n) out[i] = f2bf(in[i]);
}

// in: [rows, cols] f32 row-major  ->  out: [cols, rows] bf16 (transposed)
__global__ __launch_bounds__(256)
void cast_transpose_kernel(const float* __restrict__ in, unsigned short* __restrict__ out,
                           int rows, int cols) {
    int i = blockIdx.x * 256 + threadIdx.x;
    if (i >= rows * cols) return;
    int r = i / cols;
    int c = i - r * cols;
    out[(size_t)c * rows + r] = f2bf(in[i]);
}

// ---------------- GEMM: C[M,N] f32 = A[M,K] bf16 @ Bt[N,K] bf16 ----------------
// Block tile 128x256, 8 waves as 2(M) x 4(N), each wave owns a 64x64 register
// tile (4x4 WMMA accs). A/B K-slabs of 32 staged in LDS by the TDM, double
// buffered with s_wait_tensorcnt + workgroup barriers.
__global__ __launch_bounds__(256)
void gemm_bf16_tdm_kernel(const unsigned short* __restrict__ A,
                          const unsigned short* __restrict__ Bt,
                          float* __restrict__ Cf,
                          int M, int N, int K) {
    __shared__ alignas(16) unsigned short tA[2][128 * 32];   // 2 x 8 KB
    __shared__ alignas(16) unsigned short tB[2][256 * 32];   // 2 x 16 KB

    const int wave = threadIdx.x >> 5;
    const int lane = threadIdx.x & 31;
    const int hf   = lane >> 4;        // half-wave 0/1
    const int l15  = lane & 15;
    const int wm   = wave >> 2;        // 0..1 : 64-row slab in block tile
    const int wn   = wave & 3;         // 0..3 : 64-col slab in block tile
    const int koffA = hf * 8;          // A-frag K offset (elements)
    const int koffB = hf * 8;          // B-frag K offset (elements)

    const int nTiles = N >> 8;                 // 256-wide column blocks
    const int bm = blockIdx.x / nTiles;
    const int bn = blockIdx.x - bm * nTiles;

    const unsigned short* Abase = A  + (size_t)(bm * 128) * K;
    const unsigned short* Bbase = Bt + (size_t)(bn * 256) * K;

    const unsigned ldsA[2] = { (unsigned)(size_t)&tA[0][0], (unsigned)(size_t)&tA[1][0] };
    const unsigned ldsB[2] = { (unsigned)(size_t)&tB[0][0], (unsigned)(size_t)&tB[1][0] };

    v8f acc[4][4];
#pragma unroll
    for (int i = 0; i < 4; ++i)
#pragma unroll
        for (int j = 0; j < 4; ++j)
#pragma unroll
            for (int r = 0; r < 8; ++r) acc[i][j][r] = 0.f;

    const int nk = K >> 5;
    if (wave == 0) {                       // prologue DMA: K-slab 0 -> buffer 0
        tdm_load_tile(ldsA[0], Abase, 128u, (unsigned)M, (unsigned)K);
        tdm_load_tile(ldsB[0], Bbase, 256u, (unsigned)N, (unsigned)K);
    }

    for (int ks = 0; ks < nk; ++ks) {
        const int buf = ks & 1;
        if (wave == 0) {
            __builtin_amdgcn_s_wait_tensorcnt(0);        // buffer `buf` landed
            if (ks + 1 < nk) {                           // prefetch next slab
                tdm_load_tile(ldsA[buf ^ 1], Abase + (size_t)(ks + 1) * 32,
                              128u, (unsigned)M, (unsigned)K);
                tdm_load_tile(ldsB[buf ^ 1], Bbase + (size_t)(ks + 1) * 32,
                              256u, (unsigned)N, (unsigned)K);
            }
        }
        __syncthreads();                                 // slab visible to all waves

        const unsigned short* la = &tA[buf][0];
        const unsigned short* lb = &tB[buf][0];

        FragBF af[4];
#pragma unroll
        for (int ti = 0; ti < 4; ++ti) {
            const unsigned short* rp = la + (wm * 64 + ti * 16 + l15) * 32;
            af[ti].q[0] = *(const uint4*)(rp + koffA);
            af[ti].q[1] = *(const uint4*)(rp + 16 + koffA);
        }
#pragma unroll
        for (int tj = 0; tj < 4; ++tj) {
            FragBF bfr;
            const unsigned short* rp = lb + (wn * 64 + tj * 16 + l15) * 32;
            bfr.q[0] = *(const uint4*)(rp + koffB);
            bfr.q[1] = *(const uint4*)(rp + koffB + 16);
#pragma unroll
            for (int ti = 0; ti < 4; ++ti)
                acc[ti][tj] = __builtin_amdgcn_wmma_f32_16x16x32_bf16(
                    false, af[ti].bf, false, bfr.bf, (short)0, acc[ti][tj], false, false);
        }
        __syncthreads();                                 // done reading before overwrite
    }

    // epilogue: element r -> row bm*128 + wm*64 + ti*16 + hf*8 + r,
    //                       col bn*256 + wn*64 + tj*16 + l15
#pragma unroll
    for (int ti = 0; ti < 4; ++ti) {
#pragma unroll
        for (int tj = 0; tj < 4; ++tj) {
            size_t cbase = (size_t)(bm * 128 + wm * 64 + ti * 16 + hf * 8) * N
                         + (size_t)(bn * 256 + wn * 64 + tj * 16 + l15);
#pragma unroll
            for (int r = 0; r < 8; ++r) Cf[cbase + (size_t)r * N] = acc[ti][tj][r];
        }
    }
}

// ---------------- RoPE + scatter ----------------
__global__ __launch_bounds__(256)
void rope_scatter_kernel(const float* __restrict__ qkv, const int* __restrict__ sp_p,
                         unsigned short* __restrict__ qh, unsigned short* __restrict__ kh,
                         unsigned short* __restrict__ vT,
                         float* __restrict__ kout, float* __restrict__ vout) {
    int i = blockIdx.x * 256 + threadIdx.x;       // B*T*H*64 threads
    int d = i & 63;
    int h = (i >> 6) & (H_ - 1);
    int t = (i >> 10) & (T_ - 1);
    int b = i >> 21;
    int sp = sp_p[0];

    size_t row = (size_t)(b * T_ + t) * C3_ + (size_t)h * HD_;
    float q1 = qkv[row + d],        q2 = qkv[row + d + 64];
    float k1 = qkv[row + C_ + d],   k2 = qkv[row + C_ + d + 64];
    float v1 = qkv[row + 2*C_ + d], v2 = qkv[row + 2*C_ + d + 64];

    float freq = __powf(10000.0f, -(float)(2 * d) * (1.0f / 128.0f));
    float ang = (float)(sp + t) * freq;
    float cs = __cosf(ang), sn = __sinf(ang);

    float qr1 = q1 * cs - q2 * sn, qr2 = q1 * sn + q2 * cs;
    float kr1 = k1 * cs - k2 * sn, kr2 = k1 * sn + k2 * cs;

    size_t o = ((size_t)(b * H_ + h) * T_ + t) * HD_ + d;
    qh[o] = f2bf(qr1);  qh[o + 64] = f2bf(qr2);
    kh[o] = f2bf(kr1);  kh[o + 64] = f2bf(kr2);
    kout[o] = kr1;      kout[o + 64] = kr2;
    vout[o] = v1;       vout[o + 64] = v2;

    size_t ot = ((size_t)(b * H_ + h) * HD_ + d) * T_ + t;
    vT[ot] = f2bf(v1);
    vT[ot + (size_t)64 * T_] = f2bf(v2);
}

// ---------------- Flash attention ----------------
__global__ __launch_bounds__(256)
void flash_attn_kernel(const unsigned short* __restrict__ qh,
                       const unsigned short* __restrict__ kh,
                       const unsigned short* __restrict__ vT,
                       unsigned short* __restrict__ attnb) {
    __shared__ alignas(16) unsigned short pl[8][16 * 32];   // per-wave P staging

    const int wave = threadIdx.x >> 5;
    const int lane = threadIdx.x & 31;
    const int hf   = lane >> 4;
    const int l15  = lane & 15;

    const int bh   = blockIdx.x >> 4;          // 0 .. B*H-1
    const int tile = blockIdx.x & 15;          // 128-row slab
    const int b    = bh >> 4;
    const int h    = bh & 15;
    const int q0   = tile * 128 + wave * 16;

    const unsigned short* qbase = qh + (size_t)bh * T_ * HD_;
    const unsigned short* kbase = kh + (size_t)bh * T_ * HD_;
    const unsigned short* vbase = vT + (size_t)bh * HD_ * T_;

    const int koffA = hf * 8;
    const int koffB = hf * 8;

    FragBF qf[4];
    {
        size_t qrow = (size_t)(q0 + l15) * HD_;
#pragma unroll
        for (int kc = 0; kc < 4; ++kc) {
            qf[kc].q[0] = *(const uint4*)(qbase + qrow + kc * 32 + koffA);
            qf[kc].q[1] = *(const uint4*)(qbase + qrow + kc * 32 + 16 + koffA);
        }
    }

    v8f acc[8];
#pragma unroll
    for (int c = 0; c < 8; ++c)
#pragma unroll
        for (int r = 0; r < 8; ++r) acc[c][r] = 0.f;

    float mrow[8], lrow[8];
#pragma unroll
    for (int r = 0; r < 8; ++r) { mrow[r] = -3.0e38f; lrow[r] = 0.f; }

    const float scale = 0.08838834764831845f;   // 1/sqrt(128)
    const int kend = q0 + 15;

    for (int j0 = 0; j0 <= kend; j0 += 32) {
        v8f s0, s1;
#pragma unroll
        for (int r = 0; r < 8; ++r) { s0[r] = 0.f; s1[r] = 0.f; }
#pragma unroll
        for (int kc = 0; kc < 4; ++kc) {
            FragBF kb0, kb1;
            size_t k0r = (size_t)(j0 + l15) * HD_ + kc * 32 + koffB;
            size_t k1r = (size_t)(j0 + 16 + l15) * HD_ + kc * 32 + koffB;
            kb0.q[0] = *(const uint4*)(kbase + k0r);
            kb0.q[1] = *(const uint4*)(kbase + k0r + 16);
            kb1.q[0] = *(const uint4*)(kbase + k1r);
            kb1.q[1] = *(const uint4*)(kbase + k1r + 16);
            s0 = __builtin_amdgcn_wmma_f32_16x16x32_bf16(
                false, qf[kc].bf, false, kb0.bf, (short)0, s0, false, false);
            s1 = __builtin_amdgcn_wmma_f32_16x16x32_bf16(
                false, qf[kc].bf, false, kb1.bf, (short)0, s1, false, false);
        }

#pragma unroll
        for (int r = 0; r < 8; ++r) {
            int qg = q0 + r + hf * 8;
            float a0 = s0[r] * scale;
            float a1 = s1[r] * scale;
            if (j0 + l15 > qg)      a0 = -3.0e38f;
            if (j0 + 16 + l15 > qg) a1 = -3.0e38f;
            s0[r] = a0; s1[r] = a1;
        }

        float alpha[8];
#pragma unroll
        for (int r = 0; r < 8; ++r) {
            float mx = fmaxf(s0[r], s1[r]);
#pragma unroll
            for (int o = 8; o >= 1; o >>= 1) mx = fmaxf(mx, __shfl_xor(mx, o, 16));
            float mnew = fmaxf(mrow[r], mx);
            float a  = __expf(mrow[r] - mnew);
            float p0 = __expf(s0[r] - mnew);
            float p1 = __expf(s1[r] - mnew);
            float ps = p0 + p1;
#pragma unroll
            for (int o = 8; o >= 1; o >>= 1) ps += __shfl_xor(ps, o, 16);
            lrow[r] = lrow[r] * a + ps;
            mrow[r] = mnew;
            alpha[r] = a;
            int m = r + hf * 8;
            pl[wave][m * 32 + l15]      = f2bf(p0);
            pl[wave][m * 32 + 16 + l15] = f2bf(p1);
        }

#pragma unroll
        for (int c = 0; c < 8; ++c)
#pragma unroll
            for (int r = 0; r < 8; ++r) acc[c][r] *= alpha[r];

        FragBF pa;
        pa.q[0] = *(const uint4*)&pl[wave][l15 * 32 + koffA];
        pa.q[1] = *(const uint4*)&pl[wave][l15 * 32 + 16 + koffA];
#pragma unroll
        for (int c = 0; c < 8; ++c) {
            FragBF vb;
            size_t vr = (size_t)(c * 16 + l15) * T_ + j0 + koffB;
            vb.q[0] = *(const uint4*)(vbase + vr);
            vb.q[1] = *(const uint4*)(vbase + vr + 16);
            acc[c] = __builtin_amdgcn_wmma_f32_16x16x32_bf16(
                false, pa.bf, false, vb.bf, (short)0, acc[c], false, false);
        }
    }

#pragma unroll
    for (int c = 0; c < 8; ++c) {
#pragma unroll
        for (int r = 0; r < 8; ++r) {
            float o = acc[c][r] / lrow[r];
            int qg = q0 + r + hf * 8;
            size_t idx = ((size_t)(b * T_ + qg)) * C_ + (size_t)(h * HD_ + c * 16 + l15);
            attnb[idx] = f2bf(o);
        }
    }
}

// ---------------- launcher ----------------
extern "C" void kernel_launch(void* const* d_in, const int* in_sizes, int n_in,
                              void* d_out, int out_size, void* d_ws, size_t ws_size,
                              hipStream_t stream) {
    const float* x      = (const float*)d_in[0];   // [B,T,C]
    const float* w_qkv  = (const float*)d_in[1];   // [C,3C]
    const float* w_out  = (const float*)d_in[2];   // [C,C]
    const int*   sp     = (const int*)d_in[3];

    float* y    = (float*)d_out;                       // [B,T,C]
    float* kout = y + (size_t)BT_ * C_;                // [B,H,T,hd]
    float* vout = kout + (size_t)BT_ * C_;             // [B,H,T,hd]

    char* ws = (char*)d_ws;
    unsigned short* xh    = (unsigned short*)(ws + 0);            // [BT,C] bf16
    unsigned short* wqkvT = (unsigned short*)(ws + 16777216);     // [3C,C] bf16
    unsigned short* woutT = (unsigned short*)(ws + 41943040);     // [C,C]  bf16
    unsigned short* qh    = (unsigned short*)(ws + 50331648);     // [B,H,T,hd] bf16
    unsigned short* kh    = (unsigned short*)(ws + 67108864);     // [B,H,T,hd] bf16
    unsigned short* vT    = (unsigned short*)(ws + 83886080);     // [B,H,hd,T] bf16
    unsigned short* attnb = (unsigned short*)(ws + 100663296);    // [BT,C] bf16
    float*          qkvf  = (float*)(ws + 117440512);             // [BT,3C] f32

    cast_bf16_kernel<<<(BT_ * C_) / 256, 256, 0, stream>>>(x, xh, BT_ * C_);
    cast_transpose_kernel<<<(C_ * C3_) / 256, 256, 0, stream>>>(w_qkv, wqkvT, C_, C3_);
    cast_transpose_kernel<<<(C_ * C_) / 256, 256, 0, stream>>>(w_out, woutT, C_, C_);

    // qkv = x @ w_qkv : M=4096, N=6144, K=2048 -> (4096/128)*(6144/256) = 768 blocks
    gemm_bf16_tdm_kernel<<<(BT_ / 128) * (C3_ / 256), 256, 0, stream>>>(
        xh, wqkvT, qkvf, BT_, C3_, C_);

    rope_scatter_kernel<<<(B_ * T_ * H_ * 64) / 256, 256, 0, stream>>>(
        qkvf, sp, qh, kh, vT, kout, vout);

    flash_attn_kernel<<<B_ * H_ * (T_ / 128), 256, 0, stream>>>(qh, kh, vT, attnb);

    // y = attn @ w_out : M=4096, N=2048, K=2048 -> (4096/128)*(2048/256) = 256 blocks
    gemm_bf16_tdm_kernel<<<(BT_ / 128) * (C_ / 256), 256, 0, stream>>>(
        attnb, woutT, y, BT_, C_, C_);
}
